// LinearAssocModel_3204045603497
// MI455X (gfx1250) — matrix operand order repaired and checked by reading, not compiled, not run on hardware
//
#include <hip/hip_runtime.h>

// ---------------------------------------------------------------------------
// LinearAssocModel for MI455X (gfx1250): fused encoder + linear-assoc readout.
// Compute-bound (~25.8 GFLOP, ~3MB HBM traffic) -> stream 16-token tiles
// through v_wmma_f32_16x16x32_bf16 with weights staged in LDS as bf16.
// ---------------------------------------------------------------------------

#define Hdim   64
#define VOCAB  64
#define Bsz    64
#define Lseq   8192
#define CHUNKS 8          // blocks per batch row
#define WAVES  8          // 256 threads / wave32
#define TILES_PER_WAVE 8  // 8192/16 tiles / (CHUNKS*WAVES)

typedef __attribute__((ext_vector_type(16))) __bf16 bf16x16;
typedef __attribute__((ext_vector_type(2)))  __bf16 bf16x2;
typedef __attribute__((ext_vector_type(8)))  float  f32x8;

union Frag {
    uint4   q[2];
    bf16x16 v;
};

__device__ __forceinline__ unsigned pack_bf16(float a, float b) {
    union { bf16x2 v; unsigned u; } t;
    t.v.x = (__bf16)a;
    t.v.y = (__bf16)b;
    return t.u;   // lowers to v_cvt_pk_bf16_f32
}

// A-matrix fragment (16xK row-major bf16 in LDS, stride rs elements).
// ISA 16-bit A layout: lanes 0-15 row M=l hold K kb..kb+7 then kb+16..kb+23;
// lanes 16-31 hold K kb+8..kb+15 then kb+24..kb+31.
__device__ __forceinline__ Frag load_a(const __bf16* src, int rs, int l15,
                                       int hs, int kstep) {
    Frag f;
    const __bf16* p = src + l15 * rs + kstep * 32 + hs * 8;
    f.q[0] = *(const uint4*)(p);
    f.q[1] = *(const uint4*)(p + 16);
    return f;
}

// Same A layout, but source is f32 in LDS: 4x ds_load_b128 + packed converts.
__device__ __forceinline__ Frag load_a_f32(const float* src, int rs, int l15,
                                           int hs, int kstep) {
    const float* p = src + l15 * rs + kstep * 32 + hs * 8;
    float4 f0 = *(const float4*)(p);
    float4 f1 = *(const float4*)(p + 4);
    float4 f2 = *(const float4*)(p + 16);
    float4 f3 = *(const float4*)(p + 20);
    Frag fr;
    fr.q[0].x = pack_bf16(f0.x, f0.y);
    fr.q[0].y = pack_bf16(f0.z, f0.w);
    fr.q[0].z = pack_bf16(f1.x, f1.y);
    fr.q[0].w = pack_bf16(f1.z, f1.w);
    fr.q[1].x = pack_bf16(f2.x, f2.y);
    fr.q[1].y = pack_bf16(f2.z, f2.w);
    fr.q[1].z = pack_bf16(f3.x, f3.y);
    fr.q[1].w = pack_bf16(f3.z, f3.w);
    return fr;
}

// B-matrix fragment (weights stored column-major bf16: w[col*kTot + k]).
// Lane n (0-15): col = nt*16+n, K kb..kb+7 then kb+16..kb+23; lanes 16-31: K kb+8..kb+15 then kb+24..kb+31.
__device__ __forceinline__ Frag load_b(const __bf16* w, int kTot, int l15,
                                       int hs, int nt, int kstep) {
    Frag f;
    const __bf16* p = w + (nt * 16 + l15) * kTot + kstep * 32 + hs * 8;
    f.q[0] = *(const uint4*)(p);
    f.q[1] = *(const uint4*)(p + 16);
    return f;
}

__device__ __forceinline__ f32x8 wmma_bf16(Frag a, Frag b, f32x8 c) {
    return __builtin_amdgcn_wmma_f32_16x16x32_bf16(
        false, a.v, false, b.v, (short)0, c, false, false);
}

// Sum across the 16 lanes of each half-wave (rows live per-half in C layout).
__device__ __forceinline__ float half_reduce(float x) {
    x += __shfl_xor(x, 1, 32);
    x += __shfl_xor(x, 2, 32);
    x += __shfl_xor(x, 4, 32);
    x += __shfl_xor(x, 8, 32);
    return x;
}

// ---------------------------------------------------------------------------
// Kernel 1: q[b] = encoder(token at position L-1). 64 tokens total — scalar.
// ---------------------------------------------------------------------------
__global__ void __launch_bounds__(128) assoc_q(
    const int* __restrict__ seq, const float* __restrict__ embed,
    const float* __restrict__ W1, const float* __restrict__ b1,
    const float* __restrict__ W2, const float* __restrict__ b2,
    const float* __restrict__ ln_g, const float* __restrict__ ln_b,
    float* __restrict__ qv) {
    __shared__ float se[Hdim], sa[2 * Hdim], sx[Hdim];
    const int b = blockIdx.x, t = threadIdx.x;
    const int tok = seq[b * Lseq + (Lseq - 1)];
    if (t < Hdim) se[t] = embed[tok * Hdim + t];
    __syncthreads();
    float a = b1[t];
    for (int h = 0; h < Hdim; ++h) a += se[h] * W1[h * 2 * Hdim + t];
    sa[t] = fmaxf(a, 0.f);
    __syncthreads();
    if (t < Hdim) {
        float x = se[t] + b2[t];
        for (int k = 0; k < 2 * Hdim; ++k) x += sa[k] * W2[k * Hdim + t];
        sx[t] = x;
    }
    __syncthreads();
    if (t < Hdim) {
        float mu = 0.f;
        for (int h = 0; h < Hdim; ++h) mu += sx[h];
        mu *= (1.f / Hdim);
        float var = 0.f;
        for (int h = 0; h < Hdim; ++h) { float d = sx[h] - mu; var += d * d; }
        var *= (1.f / Hdim);
        qv[b * Hdim + t] =
            (sx[t] - mu) * rsqrtf(var + 1e-5f) * ln_g[t] + ln_b[t];
    }
}

// ---------------------------------------------------------------------------
// Kernel 2: main WMMA pipeline. Grid (CHUNKS, B), 256 threads = 8 waves.
// Each wave: 8 tiles of 16 tokens; per tile 48x v_wmma_f32_16x16x32_bf16.
// ---------------------------------------------------------------------------
__global__ void __launch_bounds__(256, 1) assoc_main(
    const int* __restrict__ seq, const float* __restrict__ embed,
    const float* __restrict__ W1, const float* __restrict__ b1,
    const float* __restrict__ W2, const float* __restrict__ b2,
    const float* __restrict__ ln_g, const float* __restrict__ ln_b,
    const float* __restrict__ Wk, const float* __restrict__ bk,
    const float* __restrict__ Wv, const float* __restrict__ bv,
    const float* __restrict__ qv, float* __restrict__ partial) {
    extern __shared__ char smem[];
    // Weights, bf16 column-major (col * K + k) for direct B-fragment loads.
    __bf16* w1c = (__bf16*)smem;          // [128 cols][64 k]
    __bf16* w2c = w1c + 128 * 64;         // [64 cols][128 k]
    __bf16* wkc = w2c + 64 * 128;         // [64][64]
    __bf16* wvc = wkc + 64 * 64;          // [64][64]
    char* pw = smem + (128 * 64 + 64 * 128 + 64 * 64 + 64 * 64) * 2; // 49152B
    const int tid  = threadIdx.x;
    const int wave = tid >> 5;
    const int lane = tid & 31;
    const int l15  = lane & 15;
    const int hs   = (lane >> 4) & 1;     // half-wave select
    char* wb = pw + wave * 10240;         // per-wave scratch, 10KB
    float*  e_f   = (float*)(wb);                   // 16x64 f32  (4KB)
    __bf16* a1_bf = (__bf16*)(wb + 4096);           // 16x128 bf16 (4KB)
    __bf16* h_bf  = (__bf16*)(wb + 4096 + 4096);    // 16x64 bf16 (2KB)

    // Stage weights f32 -> bf16 col-major in LDS (one-time, coalesced reads).
    for (int i = tid; i < 64 * 128; i += 256) {
        int h = i >> 7, n = i & 127;
        w1c[n * 64 + h] = (__bf16)W1[i];
    }
    for (int i = tid; i < 128 * 64; i += 256) {
        int k = i >> 6, n = i & 63;
        w2c[n * 128 + k] = (__bf16)W2[i];
    }
    for (int i = tid; i < 64 * 64; i += 256) {
        int h = i >> 6, n = i & 63;
        wkc[n * 64 + h] = (__bf16)Wk[i];
        wvc[n * 64 + h] = (__bf16)Wv[i];
    }
    __syncthreads();

    const int b = blockIdx.y;
    const int c = blockIdx.x;

    // Per-lane loop-invariant vectors (biases / LN / q at this lane's columns).
    float b1r[8];
#pragma unroll
    for (int nt = 0; nt < 8; ++nt) b1r[nt] = b1[nt * 16 + l15];
    float b2r[4], bkr[4], bvr[4], gr[4], lbr[4], qr[4];
#pragma unroll
    for (int nt = 0; nt < 4; ++nt) {
        const int n = nt * 16 + l15;
        b2r[nt] = b2[n]; bkr[nt] = bk[n]; bvr[nt] = bv[n];
        gr[nt] = ln_g[n]; lbr[nt] = ln_b[n];
        qr[nt] = qv[b * Hdim + n];
    }
    float ctxacc[4] = {0.f, 0.f, 0.f, 0.f};

    for (int it = 0; it < TILES_PER_WAVE; ++it) {
        const int tile = c * (WAVES * TILES_PER_WAVE) + it * WAVES + wave;
        const int l0 = tile * 16;

        // Gather 16 embedding rows -> LDS (single f32 copy; bf16 A-fragments
        // are built from it with packed converts).
        const int tok_l = seq[b * Lseq + l0 + l15];
#pragma unroll
        for (int r = 0; r < 16; ++r) {
            const int tk = __shfl(tok_l, r, 32);
            e_f[r * 64 + lane]      = embed[tk * Hdim + lane];
            e_f[r * 64 + 32 + lane] = embed[tk * Hdim + 32 + lane];
        }

        // ---- GEMM 1: relu(e @ W1 + b1), 16x128, K=64 -> 16 WMMA ----
        Frag ae[2];
        ae[0] = load_a_f32(e_f, 64, l15, hs, 0);
        ae[1] = load_a_f32(e_f, 64, l15, hs, 1);
#pragma unroll
        for (int nt = 0; nt < 8; ++nt) {
            f32x8 acc = {};
#pragma unroll
            for (int ks = 0; ks < 2; ++ks)
                acc = wmma_bf16(ae[ks], load_b(w1c, 64, l15, hs, nt, ks), acc);
            acc += b1r[nt];
#pragma unroll
            for (int j = 0; j < 8; ++j)
                a1_bf[(j + hs * 8) * 128 + nt * 16 + l15] =
                    (__bf16)fmaxf(acc[j], 0.f);
        }

        // ---- GEMM 2: a1 @ W2 + b2 + e (residual), 16x64, K=128 -> 16 WMMA --
        f32x8 xt[4];
        {
            Frag aa[4];
#pragma unroll
            for (int ks = 0; ks < 4; ++ks)
                aa[ks] = load_a(a1_bf, 128, l15, hs, ks);
#pragma unroll
            for (int nt = 0; nt < 4; ++nt) {
                f32x8 acc = {};
#pragma unroll
                for (int ks = 0; ks < 4; ++ks)
                    acc = wmma_bf16(aa[ks], load_b(w2c, 128, l15, hs, nt, ks),
                                    acc);
                acc += b2r[nt];
#pragma unroll
                for (int j = 0; j < 8; ++j)
                    acc[j] += e_f[(j + hs * 8) * 64 + nt * 16 + l15];
                xt[nt] = acc;
            }
        }

        // ---- LayerNorm across H=64 (rows live per half-wave in C layout) ---
        float mu[8], rstd[8];
#pragma unroll
        for (int j = 0; j < 8; ++j) {
            float s = xt[0][j] + xt[1][j] + xt[2][j] + xt[3][j];
            mu[j] = half_reduce(s) * (1.f / 64.f);
        }
#pragma unroll
        for (int j = 0; j < 8; ++j) {
            float d0 = xt[0][j] - mu[j], d1 = xt[1][j] - mu[j];
            float d2 = xt[2][j] - mu[j], d3 = xt[3][j] - mu[j];
            float vs = half_reduce(d0 * d0 + d1 * d1 + d2 * d2 + d3 * d3);
            rstd[j] = rsqrtf(vs * (1.f / 64.f) + 1e-5f);
        }
#pragma unroll
        for (int nt = 0; nt < 4; ++nt)
#pragma unroll
            for (int j = 0; j < 8; ++j) {
                float hv = (xt[nt][j] - mu[j]) * rstd[j] * gr[nt] + lbr[nt];
                h_bf[(j + hs * 8) * 64 + nt * 16 + l15] = (__bf16)hv;
            }

        // ---- GEMM 3: k = h@Wk+bk (8 WMMA), then coeff, freeing k early ----
        Frag ah[2];
        ah[0] = load_a(h_bf, 64, l15, hs, 0);
        ah[1] = load_a(h_bf, 64, l15, hs, 1);
        float cf[8];
        {
            f32x8 kt[4];
#pragma unroll
            for (int nt = 0; nt < 4; ++nt) {
                f32x8 ak = {};
#pragma unroll
                for (int ks = 0; ks < 2; ++ks)
                    ak = wmma_bf16(ah[ks], load_b(wkc, 64, l15, hs, nt, ks),
                                   ak);
                kt[nt] = ak + bkr[nt];
            }
#pragma unroll
            for (int j = 0; j < 8; ++j) {
                float nk = 0.f, dq = 0.f;
#pragma unroll
                for (int nt = 0; nt < 4; ++nt) {
                    nk += kt[nt][j] * kt[nt][j];
                    dq += kt[nt][j] * qr[nt];
                }
                nk = half_reduce(nk);
                dq = half_reduce(dq);
                float c0 = dq / (nk + 1e-6f);
                const int row = j + hs * 8;
                cf[j] = (l0 + row == Lseq - 1) ? 0.f : c0;  // drop final token
            }
        }

        // ---- GEMM 4: v = h@Wv+bv (8 WMMA), folded straight into ctx ----
#pragma unroll
        for (int nt = 0; nt < 4; ++nt) {
            f32x8 av = {};
#pragma unroll
            for (int ks = 0; ks < 2; ++ks)
                av = wmma_bf16(ah[ks], load_b(wvc, 64, l15, hs, nt, ks), av);
            av += bvr[nt];
            float s = 0.f;
#pragma unroll
            for (int j = 0; j < 8; ++j) s += cf[j] * av[j];
            ctxacc[nt] += s;
        }
    }

    // Combine the two half-wave row groups, write per-wave partial ctx.
#pragma unroll
    for (int nt = 0; nt < 4; ++nt)
        ctxacc[nt] += __shfl_xor(ctxacc[nt], 16, 32);
    if (lane < 16) {
#pragma unroll
        for (int nt = 0; nt < 4; ++nt)
            partial[((b * CHUNKS + c) * WAVES + wave) * Hdim + nt * 16 + lane] =
                ctxacc[nt];
    }
}

// ---------------------------------------------------------------------------
// Kernel 3: reduce partials (fixed order -> deterministic), apply Wrp & Wo.
// ---------------------------------------------------------------------------
__global__ void __launch_bounds__(64) assoc_out(
    const float* __restrict__ partial, const float* __restrict__ Wrp,
    const float* __restrict__ brp, const float* __restrict__ Wo,
    const float* __restrict__ bo, float* __restrict__ out) {
    __shared__ float sc[Hdim], sr[Hdim];
    const int b = blockIdx.x, t = threadIdx.x;
    float cs = 0.f;
    for (int p = 0; p < CHUNKS * WAVES; ++p)
        cs += partial[(b * CHUNKS * WAVES + p) * Hdim + t];
    sc[t] = cs;
    __syncthreads();
    float rp = brp[t];
    for (int h = 0; h < Hdim; ++h) rp += sc[h] * Wrp[h * Hdim + t];
    sr[t] = rp;
    __syncthreads();
    float o = bo[t];
    for (int j = 0; j < Hdim; ++j) o += sr[j] * Wo[j * VOCAB + t];
    out[b * VOCAB + t] = o;
}

// ---------------------------------------------------------------------------
extern "C" void kernel_launch(void* const* d_in, const int* in_sizes, int n_in,
                              void* d_out, int out_size, void* d_ws,
                              size_t ws_size, hipStream_t stream) {
    const int*   seq   = (const int*)d_in[0];
    const float* embed = (const float*)d_in[1];
    const float* W1    = (const float*)d_in[2];
    const float* b1    = (const float*)d_in[3];
    const float* W2    = (const float*)d_in[4];
    const float* b2    = (const float*)d_in[5];
    const float* ln_g  = (const float*)d_in[6];
    const float* ln_b  = (const float*)d_in[7];
    const float* Wk    = (const float*)d_in[8];
    const float* bk    = (const float*)d_in[9];
    const float* Wv    = (const float*)d_in[10];
    const float* bv    = (const float*)d_in[11];
    const float* Wrp   = (const float*)d_in[12];
    const float* brp   = (const float*)d_in[13];
    const float* Wo    = (const float*)d_in[14];
    const float* bo    = (const float*)d_in[15];
    float* outp = (float*)d_out;

    float* qv      = (float*)d_ws;                 // B*H floats
    float* partial = qv + Bsz * Hdim;              // B*CHUNKS*WAVES*H floats

    assoc_q<<<Bsz, 128, 0, stream>>>(seq, embed, W1, b1, W2, b2, ln_g, ln_b,
                                     qv);

    const size_t shmem = 49152 + WAVES * 10240;    // weights + wave scratch
    assoc_main<<<dim3(CHUNKS, Bsz), 256, shmem, stream>>>(
        seq, embed, W1, b1, W2, b2, ln_g, ln_b, Wk, bk, Wv, bv, qv, partial);

    assoc_out<<<Bsz, 64, 0, stream>>>(partial, Wrp, brp, Wo, bo, outp);
}
